// EncodeProcessDecode3_59339268162142
// MI455X (gfx1250) — compile-verified, matrix operand order checked by reading, not run on hardware
//
#include <hip/hip_runtime.h>
#include <hip/hip_bf16.h>
#include <stdint.h>

// ---------------------------------------------------------------------------
// CDNA5 (gfx1250) wave32 WMMA GNN pipeline (EncodeProcessDecode).
// Matrix work: v_wmma_f32_16x16x32_bf16 (bf16 in, fp32 accum).
// Each wave computes TWO 16x16 tiles stacked in M (B-fragment reuse, 2x ILP).
// ---------------------------------------------------------------------------

typedef __attribute__((ext_vector_type(16))) __bf16 v16bf;
typedef __attribute__((ext_vector_type(8)))  float  v8f;

#define LN_EPS 1e-5f

// Load a per-lane 16-element bf16 WMMA fragment slice.
// ISA 16-bit A layout (16x32): lanes 0-15 hold K {k0..k0+7, k0+16..k0+23},
// lanes 16-31 hold K {k0+8..k0+15, k0+24..k0+31}. B (col-major weights) mirrors it.
__device__ __forceinline__ v16bf load_frag(const __bf16* base, int k0, int hsel) {
  union { uint4 q[2]; v16bf v; } u;
  const __bf16* p = base + k0 + hsel * 8;
  u.q[0] = *reinterpret_cast<const uint4*>(p);
  u.q[1] = *reinterpret_cast<const uint4*>(p + 16);
  return u.v;
}

// ------------------------------- utility kernels ---------------------------

__global__ void k_zero_f32(float* p, long n) {
  long i = (long)blockIdx.x * blockDim.x + threadIdx.x;
  if (i < n) p[i] = 0.f;
}

__global__ void k_copy_f32(float* dst, const float* src, int n) {
  int i = blockIdx.x * blockDim.x + threadIdx.x;
  if (i < n) dst[i] = src[i];
}

// fp32 row-major W[K x N] -> bf16 col-major Wt[N x K]
__global__ void k_transpose_bf16(const float* __restrict__ W, __bf16* __restrict__ Wt,
                                 int K, int N) {
  long i = (long)blockIdx.x * blockDim.x + threadIdx.x;
  if (i >= (long)K * N) return;
  int k = (int)(i / N), n = (int)(i % N);
  Wt[(size_t)n * K + k] = (__bf16)W[i];
}

// tiny-K dense layer (encoders, K<=4): h = relu(A W1 + b1), N fixed 128
__global__ void k_dense_small(const float* __restrict__ A, int M, int K,
                              const float* __restrict__ W, const float* __restrict__ b,
                              __bf16* __restrict__ H) {
  long i = (long)blockIdx.x * blockDim.x + threadIdx.x;
  if (i >= (long)M * 128) return;
  int m = (int)(i >> 7), n = (int)(i & 127);
  float s = b[n];
  for (int k = 0; k < K; ++k) s += A[(size_t)m * K + k] * W[k * 128 + n];
  H[i] = (__bf16)(s > 0.f ? s : 0.f);
}

// ------------------------------- WMMA GEMM ---------------------------------
// Y = act(A[MxK](bf16) * Wt[col-major 128xK](bf16) + bias), optional LayerNorm.
// 256 threads = 8 waves; each wave computes one 16-col tile x TWO 16-row tiles;
// block covers 32 rows x 128 cols.
__global__ __launch_bounds__(256)
void k_gemm_bf16(const __bf16* __restrict__ A, int sA, int M, int K,
                 const __bf16* __restrict__ Wt, const float* __restrict__ bias,
                 int do_ln, const float* __restrict__ g, const float* __restrict__ lb,
                 __bf16* __restrict__ out0, int so0,
                 __bf16* __restrict__ out1, int so1) {
  __shared__ float tile[32 * 128];
  __shared__ float mu[32], rs[32];
  const int tid = threadIdx.x;
  const int w = tid >> 5, lane = tid & 31, hsel = lane >> 4, lr = lane & 15;
  const int m0 = blockIdx.x * 32;
  int r0 = m0 + lr;       if (r0 > M - 1) r0 = M - 1;
  int r1 = m0 + 16 + lr;  if (r1 > M - 1) r1 = M - 1;
  const __bf16* Ap0 = A + (size_t)r0 * sA;
  const __bf16* Ap1 = A + (size_t)r1 * sA;
  const __bf16* Bp = Wt + (size_t)(w * 16 + lr) * K;
  v8f c0 = {}, c1 = {};
  for (int k0 = 0; k0 < K; k0 += 32) {
    v16bf b  = load_frag(Bp, k0, hsel);
    v16bf a0 = load_frag(Ap0, k0, hsel);
    v16bf a1 = load_frag(Ap1, k0, hsel);
    c0 = __builtin_amdgcn_wmma_f32_16x16x32_bf16(false, a0, false, b, (short)0, c0, false, false);
    c1 = __builtin_amdgcn_wmma_f32_16x16x32_bf16(false, a1, false, b, (short)0, c1, false, false);
  }
  const int gc = w * 16 + lr;
#pragma unroll
  for (int r = 0; r < 8; ++r) {
    float v0 = c0[r] + bias[gc];
    float v1 = c1[r] + bias[gc];
    v0 = v0 > 0.f ? v0 : 0.f;              // act_final (all users have ReLU)
    v1 = v1 > 0.f ? v1 : 0.f;
    tile[(r + hsel * 8) * 128 + gc] = v0;
    tile[(16 + r + hsel * 8) * 128 + gc] = v1;
  }
  __syncthreads();
  if (do_ln) {
    if (tid < 32) {
      float s = 0.f, s2 = 0.f;
      for (int i = 0; i < 128; ++i) { float v = tile[tid * 128 + i]; s += v; s2 += v * v; }
      float m = s * (1.f / 128.f);
      float var = s2 * (1.f / 128.f) - m * m;
      mu[tid] = m; rs[tid] = rsqrtf(var + LN_EPS);
    }
    __syncthreads();
  }
  for (int j = 0; j < 16; ++j) {
    int idx = tid + j * 256;
    int rl = idx >> 7, cc = idx & 127;
    int grow = m0 + rl;
    if (grow < M) {
      float v = tile[idx];
      if (do_ln) v = (v - mu[rl]) * rs[rl] * g[cc] + lb[cc];
      __bf16 bv = (__bf16)v;
      out0[(size_t)grow * so0 + cc] = bv;
      if (out1) out1[(size_t)grow * so1 + cc] = bv;
    }
  }
}

// core-edge layer1: h = relu(cx[row]W1a + cx[col]W1b + ce W1c + gvec)
// gvec = cu*W1d + b1 (precomputed per step). w1t is 128x1024 col-major.
// Block covers 32 edges (2 M-tiles per wave, shared B fragments).
__global__ __launch_bounds__(256)
void k_core_edge_l1(const __bf16* __restrict__ cx, const __bf16* __restrict__ ce,
                    const int* __restrict__ rowi, const int* __restrict__ coli, int E,
                    const __bf16* __restrict__ w1t, const float* __restrict__ gvec,
                    __bf16* __restrict__ H) {
  const int tid = threadIdx.x;
  const int w = tid >> 5, lane = tid & 31, hsel = lane >> 4, lr = lane & 15;
  const int m0 = blockIdx.x * 32;
  int e0 = m0 + lr;       if (e0 > E - 1) e0 = E - 1;
  int e1 = m0 + 16 + lr;  if (e1 > E - 1) e1 = E - 1;
  const __bf16* segA[3];   // tile 0 (edges m0..m0+15)
  const __bf16* segB[3];   // tile 1 (edges m0+16..m0+31)
  segA[0] = cx + (size_t)rowi[e0] * 256;
  segA[1] = cx + (size_t)coli[e0] * 256;
  segA[2] = ce + (size_t)e0 * 256;
  segB[0] = cx + (size_t)rowi[e1] * 256;
  segB[1] = cx + (size_t)coli[e1] * 256;
  segB[2] = ce + (size_t)e1 * 256;
  const __bf16* Bp = w1t + (size_t)(w * 16 + lr) * 1024;
  v8f c0 = {}, c1 = {};
  for (int s = 0; s < 3; ++s) {
    const __bf16* Ap0 = segA[s];
    const __bf16* Ap1 = segB[s];
    const __bf16* Bs = Bp + s * 256;
    for (int k0 = 0; k0 < 256; k0 += 32) {
      v16bf b  = load_frag(Bs, k0, hsel);
      v16bf a0 = load_frag(Ap0, k0, hsel);
      v16bf a1 = load_frag(Ap1, k0, hsel);
      c0 = __builtin_amdgcn_wmma_f32_16x16x32_bf16(false, a0, false, b, (short)0, c0, false, false);
      c1 = __builtin_amdgcn_wmma_f32_16x16x32_bf16(false, a1, false, b, (short)0, c1, false, false);
    }
  }
  const int gc = w * 16 + lr;
#pragma unroll
  for (int r = 0; r < 8; ++r) {
    int g0 = m0 + r + hsel * 8;
    int g1 = m0 + 16 + r + hsel * 8;
    float v0 = c0[r] + gvec[gc];
    float v1 = c1[r] + gvec[gc];
    v0 = v0 > 0.f ? v0 : 0.f;
    v1 = v1 > 0.f ? v1 : 0.f;
    if (g0 < E) H[(size_t)g0 * 128 + gc] = (__bf16)v0;
    if (g1 < E) H[(size_t)g1 * 128 + gc] = (__bf16)v1;
  }
}

// core-node layer1: h = relu(cx W1a + agg W1b + gvec). w1t is 128x640 col-major.
// Block covers 32 nodes (2 M-tiles per wave).
__global__ __launch_bounds__(256)
void k_core_node_l1(const __bf16* __restrict__ cx, const __bf16* __restrict__ agg, int M,
                    const __bf16* __restrict__ w1t, const float* __restrict__ gvec,
                    __bf16* __restrict__ H) {
  const int tid = threadIdx.x;
  const int w = tid >> 5, lane = tid & 31, hsel = lane >> 4, lr = lane & 15;
  const int m0 = blockIdx.x * 32;
  int r0 = m0 + lr;       if (r0 > M - 1) r0 = M - 1;
  int r1 = m0 + 16 + lr;  if (r1 > M - 1) r1 = M - 1;
  const __bf16* x0 = cx + (size_t)r0 * 256;
  const __bf16* x1 = cx + (size_t)r1 * 256;
  const __bf16* g0p = agg + (size_t)r0 * 128;
  const __bf16* g1p = agg + (size_t)r1 * 128;
  const __bf16* Bp = w1t + (size_t)(w * 16 + lr) * 640;
  v8f c0 = {}, c1 = {};
  for (int k0 = 0; k0 < 256; k0 += 32) {
    v16bf b  = load_frag(Bp, k0, hsel);
    v16bf a0 = load_frag(x0, k0, hsel);
    v16bf a1 = load_frag(x1, k0, hsel);
    c0 = __builtin_amdgcn_wmma_f32_16x16x32_bf16(false, a0, false, b, (short)0, c0, false, false);
    c1 = __builtin_amdgcn_wmma_f32_16x16x32_bf16(false, a1, false, b, (short)0, c1, false, false);
  }
  for (int k0 = 0; k0 < 128; k0 += 32) {
    v16bf b  = load_frag(Bp + 256, k0, hsel);
    v16bf a0 = load_frag(g0p, k0, hsel);
    v16bf a1 = load_frag(g1p, k0, hsel);
    c0 = __builtin_amdgcn_wmma_f32_16x16x32_bf16(false, a0, false, b, (short)0, c0, false, false);
    c1 = __builtin_amdgcn_wmma_f32_16x16x32_bf16(false, a1, false, b, (short)0, c1, false, false);
  }
  const int gc = w * 16 + lr;
#pragma unroll
  for (int r = 0; r < 8; ++r) {
    int gr0 = m0 + r + hsel * 8;
    int gr1 = m0 + 16 + r + hsel * 8;
    float v0 = c0[r] + gvec[gc];
    float v1 = c1[r] + gvec[gc];
    v0 = v0 > 0.f ? v0 : 0.f;
    v1 = v1 > 0.f ? v1 : 0.f;
    if (gr0 < M) H[(size_t)gr0 * 128 + gc] = (__bf16)v0;
    if (gr1 < M) H[(size_t)gr1 * 128 + gc] = (__bf16)v1;
  }
}

// -------------------------- scatter-mean (atomics) -------------------------

__global__ void k_count(const int* __restrict__ col, int E, float* cnt) {
  int e = blockIdx.x * blockDim.x + threadIdx.x;
  if (e < E) atomicAdd(&cnt[col[e]], 1.0f);
}

__global__ void k_scatter(const __bf16* __restrict__ eC, const int* __restrict__ col,
                          int E, float* __restrict__ acc) {
  long i = (long)blockIdx.x * blockDim.x + threadIdx.x;
  if (i >= (long)E * 128) return;
  int e = (int)(i >> 7), f = (int)(i & 127);
  atomicAdd(&acc[(size_t)col[e] * 128 + f], (float)eC[i]);
}

__global__ void k_aggfin(const float* __restrict__ acc, const float* __restrict__ cnt,
                         int N, __bf16* __restrict__ agg) {
  long i = (long)blockIdx.x * blockDim.x + threadIdx.x;
  if (i >= (long)N * 128) return;
  int n = (int)(i >> 7);
  float c = cnt[n]; c = c > 1.f ? c : 1.f;
  agg[i] = (__bf16)(acc[i] / c);
}

// column means: block-partial sums + one atomicAdd per feature per block
__global__ void k_colmean(const __bf16* __restrict__ A, int M, int rpb, float* out) {
  int f = threadIdx.x;  // 128
  int r0 = blockIdx.x * rpb;
  int r1 = r0 + rpb; if (r1 > M) r1 = M;
  float s = 0.f;
  for (int r = r0; r < r1; ++r) s += (float)A[(size_t)r * 128 + f];
  atomicAdd(&out[f], s);
}

// ------------------- single-row GEMV (global MLPs, fp32) -------------------
// out[n] = act(sum over up-to-4 segments s_i (scaled) * W[k,n] + b[n]); W row-major
__global__ void k_gemv(const float* s0, int l0, float c0,
                       const float* s1, int l1, float c1,
                       const float* s2, int l2, float c2,
                       const float* s3, int l3, float c3,
                       const float* __restrict__ W, int N,
                       const float* __restrict__ bias, int relu,
                       float* __restrict__ out) {
  int n = threadIdx.x;
  if (n >= N) return;
  float acc = bias[n];
  const float* segs[4] = { s0, s1, s2, s3 };
  int lens[4] = { l0, l1, l2, l3 };
  float cs[4] = { c0, c1, c2, c3 };
  int kg = 0;
  for (int s = 0; s < 4; ++s) {
    const float* sp = segs[s];
    for (int k = 0; k < lens[s]; ++k)
      acc += sp[k] * cs[s] * W[(size_t)(kg + k) * N + n];
    kg += lens[s];
  }
  if (relu && acc < 0.f) acc = 0.f;
  out[n] = acc;
}

// ----------------------- output heads (VALU, small N) ----------------------

__global__ void k_out_l1(const __bf16* __restrict__ A, int sA, int M, int K,
                         const float* __restrict__ W, const float* __restrict__ b,
                         int N, __bf16* __restrict__ H) {
  long i = (long)blockIdx.x * blockDim.x + threadIdx.x;
  if (i >= (long)M * N) return;
  int m = (int)(i / N), n = (int)(i % N);
  float s = b[n];
  const __bf16* a = A + (size_t)m * sA;
  for (int k = 0; k < K; ++k) s += (float)a[k] * W[k * N + n];
  H[(size_t)m * N + n] = (__bf16)(s > 0.f ? s : 0.f);
}

__global__ void k_out_l2(const __bf16* __restrict__ H, int M, int K,
                         const float* __restrict__ W, const float* __restrict__ b,
                         int N, float* __restrict__ out) {
  long i = (long)blockIdx.x * blockDim.x + threadIdx.x;
  if (i >= (long)M * N) return;
  int m = (int)(i / N), n = (int)(i % N);
  float s = b[n];
  const __bf16* h = H + (size_t)m * K;
  for (int k = 0; k < K; ++k) s += (float)h[k] * W[k * N + n];
  out[(size_t)m * N + n] = s;
}

// ---------------------------------------------------------------------------

extern "C" void kernel_launch(void* const* d_in, const int* in_sizes, int n_in,
                              void* d_out, int out_size, void* d_ws, size_t ws_size,
                              hipStream_t stream) {
  (void)n_in; (void)out_size; (void)ws_size;
  const float* X  = (const float*)d_in[0];
  const int*   EI = (const int*)d_in[1];
  const float* EA = (const float*)d_in[2];
  const float* GA = (const float*)d_in[3];
  const int Nn = in_sizes[0] / 4;
  const int E  = in_sizes[1] / 2;
  const int* rowI = EI;
  const int* colI = EI + E;

  // params flattened in JAX pytree order (dicts sorted alphabetically)
#define P(i) ((const float*)d_in[(i)])
  // core_edge: b1=4 b2=5 ln_b=6 ln_g=7 w1=8 w2=9
  // core_glob: b1=10 b2=11 w1=12 w2=13
  // core_node: b1=14 b2=15 ln_b=16 ln_g=17 w1=18 w2=19
  // dec_edge:  b1=20 b2=21 ln_b=22 ln_g=23 w1=24 w2=25
  // dec_glob:  b1=26 b2=27 w1=28 w2=29
  // dec_node:  b1=30 b2=31 ln_b=32 ln_g=33 w1=34 w2=35
  // enc_edge:  b1=36 b2=37 ln_b=38 ln_g=39 w1=40 w2=41
  // enc_glob:  b1=42 b2=43 w1=44 w2=45
  // enc_node:  b1=46 b2=47 ln_b=48 ln_g=49 w1=50 w2=51
  // out_edge:  b1=52 b2=53 w1=54 w2=55
  // out_glob:  b1=56 b2=57 w1=58 w2=59
  // out_node:  b1=60 b2=61 w1=62 w2=63

  char* base = (char*)d_ws;
  size_t off = 0;
  auto alloc = [&](size_t bytes) -> void* {
    off = (off + 255) & ~(size_t)255;
    void* p = base + off;
    off += bytes;
    return p;
  };

  __bf16* wt_en_w2 = (__bf16*)alloc(16384 * 2);
  __bf16* wt_ee_w2 = (__bf16*)alloc(16384 * 2);
  __bf16* wt_ce_w1 = (__bf16*)alloc(131072 * 2);
  __bf16* wt_ce_w2 = (__bf16*)alloc(16384 * 2);
  __bf16* wt_cn_w1 = (__bf16*)alloc(81920 * 2);
  __bf16* wt_cn_w2 = (__bf16*)alloc(16384 * 2);
  __bf16* wt_dn_w1 = (__bf16*)alloc(16384 * 2);
  __bf16* wt_dn_w2 = (__bf16*)alloc(16384 * 2);
  __bf16* wt_de_w1 = (__bf16*)alloc(16384 * 2);
  __bf16* wt_de_w2 = (__bf16*)alloc(16384 * 2);
  __bf16* cx   = (__bf16*)alloc((size_t)Nn * 256 * 2);
  __bf16* ce   = (__bf16*)alloc((size_t)E * 256 * 2);
  __bf16* hE   = (__bf16*)alloc((size_t)E * 128 * 2);
  __bf16* eC   = (__bf16*)alloc((size_t)E * 128 * 2);
  __bf16* hN   = (__bf16*)alloc((size_t)Nn * 128 * 2);
  __bf16* xC   = (__bf16*)alloc((size_t)Nn * 128 * 2);
  __bf16* agg  = (__bf16*)alloc((size_t)Nn * 128 * 2);
  __bf16* hOut = (__bf16*)alloc((size_t)E * 65 * 2);
  float* acc   = (float*)alloc((size_t)Nn * 128 * 4);
  float* cnt   = (float*)alloc((size_t)Nn * 4);
  float* u0f   = (float*)alloc(128 * 4);
  float* uf    = (float*)alloc(128 * 4);
  float* ug    = (float*)alloc(128 * 4);
  float* hg    = (float*)alloc(128 * 4);
  float* means = (float*)alloc(256 * 4);   // xmean | emean
  float* xmean = means, *emean = means + 128;
  float* gve   = (float*)alloc(128 * 4);
  float* gvn   = (float*)alloc(128 * 4);

  auto T = [&](const float* W, __bf16* Wt, int K, int N) {
    long tot = (long)K * N;
    k_transpose_bf16<<<(unsigned)((tot + 255) / 256), 256, 0, stream>>>(W, Wt, K, N);
  };
  auto GEMM = [&](const __bf16* A, int sA, int M, int K, const __bf16* Wt, const float* b,
                  int ln, const float* g, const float* lb,
                  __bf16* o0, int s0, __bf16* o1, int s1) {
    k_gemm_bf16<<<(unsigned)((M + 31) / 32), 256, 0, stream>>>(A, sA, M, K, Wt, b, ln, g, lb, o0, s0, o1, s1);
  };
  auto GEMV = [&](const float* s0, int l0, float c0, const float* s1, int l1, float c1,
                  const float* s2, int l2, float c2, const float* s3, int l3, float c3,
                  const float* W, int N, const float* b, int relu, float* out) {
    k_gemv<<<1, 128, 0, stream>>>(s0, l0, c0, s1, l1, c1, s2, l2, c2, s3, l3, c3, W, N, b, relu, out);
  };
  auto ZERO = [&](float* p, long n) {
    k_zero_f32<<<(unsigned)((n + 255) / 256), 256, 0, stream>>>(p, n);
  };

  // ---- weight prep (bf16 col-major) ----
  T(P(51), wt_en_w2, 128, 128);
  T(P(41), wt_ee_w2, 128, 128);
  T(P(8),  wt_ce_w1, 1024, 128);
  T(P(9),  wt_ce_w2, 128, 128);
  T(P(18), wt_cn_w1, 640, 128);
  T(P(19), wt_cn_w2, 128, 128);
  T(P(34), wt_dn_w1, 128, 128);
  T(P(35), wt_dn_w2, 128, 128);
  T(P(24), wt_de_w1, 128, 128);
  T(P(25), wt_de_w2, 128, 128);

  // ---- receiver counts (constant across steps) ----
  ZERO(cnt, Nn);
  k_count<<<(unsigned)((E + 255) / 256), 256, 0, stream>>>(colI, E, cnt);

  // ---- encoders ----
  k_dense_small<<<(unsigned)(((long)Nn * 128 + 255) / 256), 256, 0, stream>>>(X, Nn, 4, P(50), P(46), hN);
  GEMM(hN, 128, Nn, 128, wt_en_w2, P(47), 1, P(49), P(48), cx, 256, cx + 128, 256);   // x0 = x
  k_dense_small<<<(unsigned)(((long)E * 128 + 255) / 256), 256, 0, stream>>>(EA, E, 3, P(40), P(36), hE);
  GEMM(hE, 128, E, 128, wt_ee_w2, P(37), 1, P(39), P(38), ce, 256, ce + 128, 256);    // e0 = e
  GEMV(GA, 2, 1.f, nullptr, 0, 0.f, nullptr, 0, 0.f, nullptr, 0, 0.f, P(44), 128, P(42), 1, hg);
  GEMV(hg, 128, 1.f, nullptr, 0, 0.f, nullptr, 0, 0.f, nullptr, 0, 0.f, P(45), 128, P(43), 1, u0f);
  k_copy_f32<<<1, 128, 0, stream>>>(uf, u0f, 128);  // u = u0

  // ---- 5 message-passing steps ----
  for (int step = 0; step < 5; ++step) {
    // per-step global contributions: gvec = cu * W1_glob_block + b1
    GEMV(u0f, 128, 1.f, uf, 128, 1.f, nullptr, 0, 0.f, nullptr, 0, 0.f,
         P(8) + (size_t)768 * 128, 128, P(4), 0, gve);
    GEMV(u0f, 128, 1.f, uf, 128, 1.f, nullptr, 0, 0.f, nullptr, 0, 0.f,
         P(18) + (size_t)384 * 128, 128, P(14), 0, gvn);

    // core edge MLP
    k_core_edge_l1<<<(unsigned)((E + 31) / 32), 256, 0, stream>>>(cx, ce, rowI, colI, E, wt_ce_w1, gve, hE);
    GEMM(hE, 128, E, 128, wt_ce_w2, P(5), 1, P(7), P(6), eC, 128, nullptr, 0);

    // scatter-mean to receivers
    ZERO(acc, (long)Nn * 128);
    k_scatter<<<(unsigned)(((long)E * 128 + 255) / 256), 256, 0, stream>>>(eC, colI, E, acc);
    k_aggfin<<<(unsigned)(((long)Nn * 128 + 255) / 256), 256, 0, stream>>>(acc, cnt, Nn, agg);

    // core node MLP
    k_core_node_l1<<<(unsigned)((Nn + 31) / 32), 256, 0, stream>>>(cx, agg, Nn, wt_cn_w1, gvn, hN);
    GEMM(hN, 128, Nn, 128, wt_cn_w2, P(15), 1, P(17), P(16), xC, 128, nullptr, 0);

    // core global MLP
    ZERO(means, 256);
    k_colmean<<<(unsigned)((Nn + 2047) / 2048), 128, 0, stream>>>(xC, Nn, 2048, xmean);
    k_colmean<<<(unsigned)((E + 2047) / 2048), 128, 0, stream>>>(eC, E, 2048, emean);
    GEMV(u0f, 128, 1.f, uf, 128, 1.f, xmean, 128, 1.f / (float)Nn, emean, 128, 1.f / (float)E,
         P(12), 128, P(10), 1, hg);
    GEMV(hg, 128, 1.f, nullptr, 0, 0.f, nullptr, 0, 0.f, nullptr, 0, 0.f, P(13), 128, P(11), 1, ug);

    // decoders
    GEMM(xC, 128, Nn, 128, wt_dn_w1, P(30), 0, nullptr, nullptr, hN, 128, nullptr, 0);
    GEMM(hN, 128, Nn, 128, wt_dn_w2, P(31), 1, P(33), P(32), cx + 128, 256, nullptr, 0);
    GEMM(eC, 128, E, 128, wt_de_w1, P(20), 0, nullptr, nullptr, hE, 128, nullptr, 0);
    GEMM(hE, 128, E, 128, wt_de_w2, P(21), 1, P(23), P(22), ce + 128, 256, nullptr, 0);
    GEMV(ug, 128, 1.f, nullptr, 0, 0.f, nullptr, 0, 0.f, nullptr, 0, 0.f, P(28), 128, P(26), 1, hg);
    GEMV(hg, 128, 1.f, nullptr, 0, 0.f, nullptr, 0, 0.f, nullptr, 0, 0.f, P(29), 128, P(27), 1, uf);
  }

  // ---- output transformers ----
  float* out = (float*)d_out;
  float* out_x = out;
  float* out_e = out + (size_t)Nn * 2;
  float* out_u = out + (size_t)Nn * 2 + (size_t)E * 3;

  k_out_l1<<<(unsigned)(((long)Nn * 65 + 255) / 256), 256, 0, stream>>>(cx + 128, 256, Nn, 128, P(62), P(60), 65, hOut);
  k_out_l2<<<(unsigned)(((long)Nn * 2 + 255) / 256), 256, 0, stream>>>(hOut, Nn, 65, P(63), P(61), 2, out_x);
  k_out_l1<<<(unsigned)(((long)E * 65 + 255) / 256), 256, 0, stream>>>(ce + 128, 256, E, 128, P(54), P(52), 65, hOut);
  k_out_l2<<<(unsigned)(((long)E * 3 + 255) / 256), 256, 0, stream>>>(hOut, E, 65, P(55), P(53), 3, out_e);
  GEMV(uf, 128, 1.f, nullptr, 0, 0.f, nullptr, 0, 0.f, nullptr, 0, 0.f, P(58), 65, P(56), 1, hg);
  GEMV(hg, 65, 1.f, nullptr, 0, 0.f, nullptr, 0, 0.f, nullptr, 0, 0.f, P(59), 1, P(57), 0, out_u);
#undef P
}
